// GAT_75325136437379
// MI455X (gfx1250) — compile-verified, matrix operand order checked by reading, not run on hardware
//
#include <hip/hip_runtime.h>
#include <hip/hip_bf16.h>

// ---- problem constants (match reference) ----
#define NN   40000
#define EE   640000
#define IDIM 64
#define HLD  128
#define NH   2          // heads; gat width per node = NH*HLD = 256

typedef __attribute__((ext_vector_type(16))) _Float16 v16h;
typedef __attribute__((ext_vector_type(8)))  _Float16 v8h;
typedef __attribute__((ext_vector_type(4)))  _Float16 v4h;
typedef __attribute__((ext_vector_type(8)))  float    v8f;
typedef __attribute__((ext_vector_type(4)))  float    v4f;

// fast silu: x * v_rcp_f32(1 + exp(-x))  (no IEEE div chain)
__device__ __forceinline__ float silu_f(float x) {
    return x * __builtin_amdgcn_rcpf(1.0f + __expf(-x));
}

__device__ __forceinline__ float warp_sum32(float s) {
    #pragma unroll
    for (int off = 16; off > 0; off >>= 1)
        s += __shfl_xor(s, off, 32);
    return s;
}

// float atomic max via int ordering tricks
__device__ __forceinline__ void atomicMaxF(float* addr, float val) {
    if (val >= 0.0f)
        atomicMax((int*)addr, __float_as_int(val));
    else
        atomicMin((unsigned int*)addr, __float_as_uint(val));
}

// =====================================================================
// f32 -> f16 cast, 4-wide
// =====================================================================
__global__ void cast_f32_f16(const float* __restrict__ src,
                             _Float16* __restrict__ dst, int n4)
{
    int i = blockIdx.x * blockDim.x + threadIdx.x;
    if (i >= n4) return;
    v4f v = *(const v4f*)&src[(size_t)i * 4];
    v4h h;
    #pragma unroll
    for (int j = 0; j < 4; ++j) h[j] = (_Float16)v[j];
    *(v4h*)&dst[(size_t)i * 4] = h;
}

// =====================================================================
// WMMA GEMM:  C[M,N] = act( A16[M,K] @ W16[K,N] + bias )
// BM=128, BN=64, BK=32; 8 waves = 4(M) x 2(N), each wave owns a
// 32x32 C tile = 2x2 v_wmma_f32_16x16x32_f16.
// A tile staged via CDNA5 async global->LDS (ASYNCcnt + s_wait_asynccnt),
// B tile via b128 load + transposed b16 stores. 40-half LDS row stride
// keeps every fragment start 16B-aligned for ds_load_b128.
// =====================================================================
#define BM 128
#define BN 64
#define BK 32
#define LDSTR 40

template<bool ACT, bool W16>
__global__ __launch_bounds__(256) void gemm_f16_wmma(
    const _Float16* __restrict__ A, const _Float16* __restrict__ W,
    const float* __restrict__ bias, float* __restrict__ C,
    _Float16* __restrict__ C16, int M, int K, int Ncols)
{
    __shared__ __attribute__((aligned(16))) _Float16 sA[BM * LDSTR];
    __shared__ __attribute__((aligned(16))) _Float16 sB[BN * LDSTR];

    const int t    = threadIdx.x;
    const int lane = t & 31;
    const int wid  = t >> 5;
    const int wm   = wid >> 1;   // 0..3
    const int wn   = wid & 1;    // 0..1
    const int m0   = blockIdx.x * BM;
    const int n0   = blockIdx.y * BN;
    const int half = lane >> 4;  // 0/1
    const int l16  = lane & 15;

    v8f acc[2][2] = {};

    for (int k0 = 0; k0 < K; k0 += BK) {
        // ---- A tile (BM x BK halfs): async global->LDS, b128 chunks ----
        #pragma unroll
        for (int i = 0; i < 2; ++i) {
            int idx8 = t + i * 256;          // 512 chunks of 8 halfs
            int r    = idx8 >> 2;            // 4 chunks per 32-half row
            int c8   = (idx8 & 3) * 8;
            int gr   = m0 + r;
            gr = (gr < M) ? gr : (M - 1);    // clamp: no divergence
            unsigned loff = (unsigned)(unsigned long long)&sA[r * LDSTR + c8];
            const _Float16* gp = &A[(size_t)gr * K + k0 + c8];
            asm volatile("global_load_async_to_lds_b128 %0, %1, off"
                         :: "v"(loff), "v"(gp) : "memory");
        }
        // ---- B tile (BK x BN halfs) -> sB[n][k] transposed ----
        {
            int kr = t >> 3;                 // 0..31
            int g8 = (t & 7) * 8;            // col group
            v8h bv = *(const v8h*)&W[(size_t)(k0 + kr) * Ncols + n0 + g8];
            #pragma unroll
            for (int j = 0; j < 8; ++j)
                sB[(g8 + j) * LDSTR + kr] = bv[j];
        }
        asm volatile("s_wait_asynccnt 0x0" ::: "memory");
        __syncthreads();

        // ---- fragments per documented 16-bit A/B VGPR layout ----
        v16h afr[2], bfr[2];
        #pragma unroll
        for (int tm = 0; tm < 2; ++tm) {
            int row = wm * 32 + tm * 16 + l16;
            v8h lo = *(const v8h*)&sA[row * LDSTR + half * 8];
            v8h hi = *(const v8h*)&sA[row * LDSTR + 16 + half * 8];
            afr[tm] = __builtin_shufflevector(lo, hi,
                0,1,2,3,4,5,6,7,8,9,10,11,12,13,14,15);
        }
        #pragma unroll
        for (int tn = 0; tn < 2; ++tn) {
            int col = wn * 32 + tn * 16 + l16;
            v8h lo = *(const v8h*)&sB[col * LDSTR + half * 8];
            v8h hi = *(const v8h*)&sB[col * LDSTR + 16 + half * 8];
            bfr[tn] = __builtin_shufflevector(lo, hi,
                0,1,2,3,4,5,6,7,8,9,10,11,12,13,14,15);
        }

        #pragma unroll
        for (int tm = 0; tm < 2; ++tm)
            #pragma unroll
            for (int tn = 0; tn < 2; ++tn)
                acc[tm][tn] = __builtin_amdgcn_wmma_f32_16x16x32_f16(
                    false, afr[tm], false, bfr[tn],
                    (short)0, acc[tm][tn], false, false);

        __syncthreads();
    }

    // ---- epilogue: lane = col (l16), VGPR v = row (v + 8*half) ----
    const bool fullTile = (m0 + BM) <= M;    // uniform across block
    #pragma unroll
    for (int tm = 0; tm < 2; ++tm) {
        #pragma unroll
        for (int tn = 0; tn < 2; ++tn) {
            int col = n0 + wn * 32 + tn * 16 + l16;
            float bv = bias[col];
            int row0 = m0 + wm * 32 + tm * 16 + half * 8;
            if (fullTile) {
                #pragma unroll
                for (int v = 0; v < 8; ++v) {
                    float xv = acc[tm][tn][v] + bv;
                    if (ACT) xv = silu_f(xv);
                    size_t off = (size_t)(row0 + v) * Ncols + col;
                    C[off] = xv;
                    if (W16) C16[off] = (_Float16)xv;
                }
            } else {
                #pragma unroll
                for (int v = 0; v < 8; ++v) {
                    if (row0 + v < M) {
                        float xv = acc[tm][tn][v] + bv;
                        if (ACT) xv = silu_f(xv);
                        size_t off = (size_t)(row0 + v) * Ncols + col;
                        C[off] = xv;
                        if (W16) C16[off] = (_Float16)xv;
                    }
                }
            }
        }
    }
}

// =====================================================================
// per-layer init: zero accumulator [N*HLD], max = -FLT_MAX, den = 0
// =====================================================================
__global__ void init_layer(float* __restrict__ accum,
                           float* __restrict__ maxb,
                           float* __restrict__ denb)
{
    int i = blockIdx.x * blockDim.x + threadIdx.x;
    if (i < NN * HLD) accum[i] = 0.0f;
    if (i < NN * NH) { maxb[i] = -3.402823466e38f; denb[i] = 0.0f; }
}

// =====================================================================
// edge scores: one wave per edge; 16B vector gathers; per-(dst,head)
// atomic segment max
// =====================================================================
__global__ __launch_bounds__(256) void edge_score(
    const float* __restrict__ g, const int* __restrict__ eidx,
    const float* __restrict__ att, float* __restrict__ sbuf,
    float* __restrict__ maxb)
{
    int warp = (blockIdx.x * blockDim.x + threadIdx.x) >> 5;
    int lane = threadIdx.x & 31;
    if (warp >= EE) return;
    int src = eidx[warp];
    int dst = eidx[EE + warp];
    const float* gs = g + (size_t)src * (NH * HLD);
    const float* gd = g + (size_t)dst * (NH * HLD);
    int c4 = lane * 4;
    #pragma unroll
    for (int h = 0; h < NH; ++h) {
        v4f vs = *(const v4f*)&gs[h * HLD + c4];
        v4f vd = *(const v4f*)&gd[h * HLD + c4];
        v4f va = *(const v4f*)&att[h * HLD + c4];
        float s = 0.0f;
        #pragma unroll
        for (int j = 0; j < 4; ++j) {
            float v = vs[j] + vd[j];
            v = (v > 0.0f) ? v : 0.2f * v;       // leaky_relu(0.2)
            s += v * va[j];
        }
        s = warp_sum32(s);
        if (lane == 0) {
            sbuf[(size_t)warp * NH + h] = s;
            atomicMaxF(&maxb[dst * NH + h], s);
        }
    }
}

// =====================================================================
// ex = exp(s - m[dst]); den[dst] += ex
// =====================================================================
__global__ void edge_exp(float* __restrict__ sbuf,
                         const int* __restrict__ eidx,
                         const float* __restrict__ maxb,
                         float* __restrict__ denb)
{
    int idx = blockIdx.x * blockDim.x + threadIdx.x;
    if (idx >= EE * NH) return;
    int e = idx >> 1, h = idx & 1;
    int dst = eidx[EE + e];
    float ex = __expf(sbuf[idx] - maxb[dst * NH + h]);
    sbuf[idx] = ex;
    atomicAdd(&denb[dst * NH + h], ex);
}

// =====================================================================
// scatter: out[dst] += mean_h( alpha[e,h] * h_src[h] ); one wave/edge
// =====================================================================
__global__ __launch_bounds__(256) void edge_agg(
    const float* __restrict__ g, const int* __restrict__ eidx,
    const float* __restrict__ sbuf, const float* __restrict__ denb,
    float* __restrict__ dest, float* __restrict__ alpha_out)
{
    int warp = (blockIdx.x * blockDim.x + threadIdx.x) >> 5;
    int lane = threadIdx.x & 31;
    if (warp >= EE) return;
    int src = eidx[warp];
    int dst = eidx[EE + warp];
    // alpha = ex * rcp(den + eps)   (v_rcp_f32, not IEEE div)
    float a0 = sbuf[(size_t)warp * NH + 0] *
               __builtin_amdgcn_rcpf(denb[dst * NH + 0] + 1e-16f);
    float a1 = sbuf[(size_t)warp * NH + 1] *
               __builtin_amdgcn_rcpf(denb[dst * NH + 1] + 1e-16f);
    const float* gs = g + (size_t)src * (NH * HLD);
    int c4 = lane * 4;
    v4f h0 = *(const v4f*)&gs[c4];
    v4f h1 = *(const v4f*)&gs[HLD + c4];
    #pragma unroll
    for (int j = 0; j < 4; ++j)
        atomicAdd(&dest[(size_t)dst * HLD + c4 + j],
                  0.5f * (a0 * h0[j] + a1 * h1[j]));
    if (alpha_out != nullptr && lane == 0)
        alpha_out[warp] = 0.5f * (a0 + a1);
}

// =====================================================================
// finalize GAT layer: h = silu(h + bias), write f32 (in place) + f16 mirror
// =====================================================================
__global__ void gat_finalize(float* __restrict__ buf,
                             const float* __restrict__ bias,
                             _Float16* __restrict__ out16)
{
    int idx = blockIdx.x * blockDim.x + threadIdx.x;
    if (idx >= NN * HLD) return;
    float v = silu_f(buf[idx] + bias[idx & (HLD - 1)]);
    buf[idx] = v;
    out16[idx] = (_Float16)v;
}

// =====================================================================
// final decode dot: out[n] = sum_c a[n,c]*w1[c] + b1; one wave/node
// =====================================================================
__global__ __launch_bounds__(256) void final_dot(
    const float* __restrict__ a, const float* __restrict__ w1,
    const float* __restrict__ b1, float* __restrict__ out)
{
    int warp = (blockIdx.x * blockDim.x + threadIdx.x) >> 5;
    int lane = threadIdx.x & 31;
    if (warp >= NN) return;
    int c4 = lane * 4;
    v4f av = *(const v4f*)&a[(size_t)warp * HLD + c4];
    v4f wv = *(const v4f*)&w1[c4];
    float s = av[0]*wv[0] + av[1]*wv[1] + av[2]*wv[2] + av[3]*wv[3];
    s = warp_sum32(s);
    if (lane == 0) out[warp] = s + b1[0];
}

// =====================================================================
extern "C" void kernel_launch(void* const* d_in, const int* in_sizes, int n_in,
                              void* d_out, int out_size, void* d_ws, size_t ws_size,
                              hipStream_t stream)
{
    (void)in_sizes; (void)n_in; (void)out_size; (void)ws_size;

    const float* x        = (const float*)d_in[0];
    const int*   edgeIdx  = (const int*)  d_in[1];
    const float* enc_w0   = (const float*)d_in[3];
    const float* enc_b0   = (const float*)d_in[4];
    const float* enc_w1   = (const float*)d_in[5];
    const float* enc_b1   = (const float*)d_in[6];
    const float* gat0_W   = (const float*)d_in[7];
    const float* gat0_bl  = (const float*)d_in[8];
    const float* gat0_att = (const float*)d_in[9];
    const float* gat0_bias= (const float*)d_in[10];
    const float* gat1_W   = (const float*)d_in[11];
    const float* gat1_bl  = (const float*)d_in[12];
    const float* gat1_att = (const float*)d_in[13];
    const float* gat1_bias= (const float*)d_in[14];
    const float* dec_w0   = (const float*)d_in[15];
    const float* dec_b0   = (const float*)d_in[16];
    const float* dec_w1   = (const float*)d_in[17];
    const float* dec_b1   = (const float*)d_in[18];
    float* out = (float*)d_out;

    // ---- workspace partition: f32 region ----
    float* bufA = (float*)d_ws;                 // N*HLD  (h0 -> gat0 acc -> d0)
    float* bufB = bufA + (size_t)NN * HLD;      // N*HLD  (h1 -> gat1 acc)
    float* bufG = bufB + (size_t)NN * HLD;      // N*NH*HLD (gat pre-act)
    float* maxb = bufG + (size_t)NN * NH * HLD; // N*NH
    float* denb = maxb + (size_t)NN * NH;       // N*NH
    float* ebuf = denb + (size_t)NN * NH;       // E*NH
    // ---- f16 region ----
    _Float16* xh   = (_Float16*)(ebuf + (size_t)EE * NH); // N*IDIM
    _Float16* a16  = xh  + (size_t)NN * IDIM;             // N*HLD mirror A
    _Float16* b16  = a16 + (size_t)NN * HLD;              // N*HLD mirror B
    _Float16* w_e0 = b16 + (size_t)NN * HLD;              // 64*128
    _Float16* w_e1 = w_e0 + IDIM * HLD;                   // 128*128
    _Float16* w_g0 = w_e1 + HLD * HLD;                    // 128*256
    _Float16* w_g1 = w_g0 + HLD * NH * HLD;               // 128*256
    _Float16* w_d0 = w_g1 + HLD * NH * HLD;               // 128*128

    const dim3 blk(256);
    const int mBlocks = (NN + BM - 1) / BM;     // 313
    const int initG   = (NN * HLD + 255) / 256;
    const int edgeWG  = (EE + 7) / 8;
    const int ehG     = (EE * NH + 255) / 256;
    auto c4g = [](int n) { return (n / 4 + 255) / 256; };

    // one-shot f16 casts (input + weights)
    cast_f32_f16<<<c4g(NN * IDIM), blk, 0, stream>>>(x, xh, NN * IDIM / 4);
    cast_f32_f16<<<c4g(IDIM * HLD), blk, 0, stream>>>(enc_w0, w_e0, IDIM * HLD / 4);
    cast_f32_f16<<<c4g(HLD * HLD), blk, 0, stream>>>(enc_w1, w_e1, HLD * HLD / 4);
    cast_f32_f16<<<c4g(HLD * NH * HLD), blk, 0, stream>>>(gat0_W, w_g0, HLD * NH * HLD / 4);
    cast_f32_f16<<<c4g(HLD * NH * HLD), blk, 0, stream>>>(gat1_W, w_g1, HLD * NH * HLD / 4);
    cast_f32_f16<<<c4g(HLD * HLD), blk, 0, stream>>>(dec_w0, w_d0, HLD * HLD / 4);

    // encoder (silu + f16 mirror)
    gemm_f16_wmma<true, true><<<dim3(mBlocks, 2), blk, 0, stream>>>(
        xh, w_e0, enc_b0, bufA, a16, NN, IDIM, HLD);
    gemm_f16_wmma<true, true><<<dim3(mBlocks, 2), blk, 0, stream>>>(
        a16, w_e1, enc_b1, bufB, b16, NN, HLD, HLD);

    // ---- GAT layer 0 ----
    gemm_f16_wmma<false, false><<<dim3(mBlocks, 4), blk, 0, stream>>>(
        b16, w_g0, gat0_bl, bufG, nullptr, NN, HLD, NH * HLD);
    init_layer<<<initG, blk, 0, stream>>>(bufA, maxb, denb);
    edge_score<<<edgeWG, blk, 0, stream>>>(bufG, edgeIdx, gat0_att, ebuf, maxb);
    edge_exp<<<ehG, blk, 0, stream>>>(ebuf, edgeIdx, maxb, denb);
    edge_agg<<<edgeWG, blk, 0, stream>>>(bufG, edgeIdx, ebuf, denb, bufA, nullptr);
    gat_finalize<<<initG, blk, 0, stream>>>(bufA, gat0_bias, a16);

    // ---- GAT layer 1 ----
    gemm_f16_wmma<false, false><<<dim3(mBlocks, 4), blk, 0, stream>>>(
        a16, w_g1, gat1_bl, bufG, nullptr, NN, HLD, NH * HLD);
    init_layer<<<initG, blk, 0, stream>>>(bufB, maxb, denb);
    edge_score<<<edgeWG, blk, 0, stream>>>(bufG, edgeIdx, gat1_att, ebuf, maxb);
    edge_exp<<<ehG, blk, 0, stream>>>(ebuf, edgeIdx, maxb, denb);
    edge_agg<<<edgeWG, blk, 0, stream>>>(bufG, edgeIdx, ebuf, denb, bufB,
                                         out + NN /* alpha.mean */);
    gat_finalize<<<initG, blk, 0, stream>>>(bufB, gat1_bias, b16);

    // decoder (no activation)
    gemm_f16_wmma<false, false><<<dim3(mBlocks, 2), blk, 0, stream>>>(
        b16, w_d0, dec_b0, bufA, nullptr, NN, HLD, HLD);
    final_dot<<<(NN + 7) / 8, blk, 0, stream>>>(bufA, dec_w1, dec_b1, out);
}
